// MultiHeadDifferentialAttention_21809843929107
// MI455X (gfx1250) — matrix operand order repaired and checked by reading, not compiled, not run on hardware
//
#include <hip/hip_runtime.h>
#include <hip/hip_bf16.h>

typedef __attribute__((ext_vector_type(16))) _Float16 v16h;
typedef __attribute__((ext_vector_type(8)))  _Float16 v8h;
typedef __attribute__((ext_vector_type(8)))  float    v8f;
typedef __attribute__((ext_vector_type(4)))  unsigned int u32x4;
typedef __attribute__((ext_vector_type(4)))  int i32x4;
typedef __attribute__((ext_vector_type(8)))  int i32x8;

#define B_ 2
#define T_ 2048
#define C_ 1024
#define H_ 16
#define D_ 64
#define NCOL_ (5*C_)   // 5120 concatenated output columns: [q1|k1|q2|k2|v]

#if defined(__gfx1250__) && __has_builtin(__builtin_amdgcn_tensor_load_to_lds)
#define USE_TDM 1
#else
#define USE_TDM 0
#endif

static __device__ __forceinline__ v16h cat8(v8h lo, v8h hi) {
  return __builtin_shufflevector(lo, hi, 0,1,2,3,4,5,6,7,8,9,10,11,12,13,14,15);
}
static __device__ __forceinline__ v8f vzero8() {
  v8f z = {0.f,0.f,0.f,0.f,0.f,0.f,0.f,0.f};
  return z;
}
#define WMMA_F16(A,Bm,Cacc) __builtin_amdgcn_wmma_f32_16x16x32_f16(false,(A),false,(Bm),(short)0,(Cacc),false,false)

#if USE_TDM
// ---------------------------------------------------------------------------
// Tensor Data Mover: DMA a 2D f16 tile (tile_d1 rows x tile_d0 elems, row
// stride stride0 elems in memory) into LDS at byte offset lds_off, inserting
// 4 DWORDs (8 halfs) of padding after every 32 DWORDs (= one 64-half row),
// which matches the 72-half padded LDS rows used below.
// Descriptor packing per CDNA5 ISA.
// ---------------------------------------------------------------------------
static __device__ __forceinline__ void tdm_load_2d_f16(
    unsigned lds_off, const void* gaddr,
    unsigned tile_d0, unsigned tile_d1, unsigned long long stride0_elems)
{
  const unsigned long long ga = (unsigned long long)(uintptr_t)gaddr;
  u32x4 g0;
  g0[0] = 1u;                                         // count=1, user descriptor
  g0[1] = lds_off;                                    // lds_addr (bytes)
  g0[2] = (unsigned)(ga & 0xffffffffu);               // global_addr[31:0]
  g0[3] = (unsigned)((ga >> 32) & 0x01ffffffu)        // global_addr[56:32]
        | (2u << 30);                                 // type = 2 ("image")
  i32x8 g1;
  g1[0] = (int)((1u << 16)                            // data_size = 2 bytes
              | (1u << 20)                            // pad_enable
              | (4u << 22)                            // pad_interval: 32 DWORDs
              | (3u << 25));                          // pad_amount: 4 DWORDs
  g1[1] = (int)((tile_d0 & 0xffffu) << 16);           // tensor_dim0 (low 16)
  g1[2] = (int)((tile_d1 & 0xffffu) << 16);           // tensor_dim1 (low 16)
  g1[3] = (int)((tile_d0 & 0xffffu) << 16);           // tile_dim0
  g1[4] = (int)(tile_d1 & 0xffffu);                   // tile_dim1 (tile_dim2 = 0)
  g1[5] = (int)(unsigned)(stride0_elems & 0xffffffffu);        // dim0 stride lo
  g1[6] = (int)(unsigned)((stride0_elems >> 32) & 0xffffu);    // dim0 stride hi
  g1[7] = 0;
  i32x4 z4 = {0,0,0,0};
#if __clang_major__ >= 23
  i32x8 z8 = {0,0,0,0,0,0,0,0};
  __builtin_amdgcn_tensor_load_to_lds(g0, g1, z4, z4, z8, 0);
#else
  __builtin_amdgcn_tensor_load_to_lds(g0, g1, z4, z4, 0);
#endif
}
#endif

// ---------------------------------------------------------------------------
// Kernel 1: fused projection GEMM. X[4096,1024] (fp32) x Wcat[1024,5120]
// (fp32) -> f16 outputs. q1/k1/q2/k2 stored [b,h,t,d]; v stored transposed
// [b,h,d,t] so it is directly in WMMA B-fragment-friendly layout for P@V.
// ---------------------------------------------------------------------------
__global__ __launch_bounds__(256) void proj_gemm_kernel(
    const float* __restrict__ x,
    const float* __restrict__ wq1, const float* __restrict__ wk1,
    const float* __restrict__ wq2, const float* __restrict__ wk2,
    const float* __restrict__ wv,
    _Float16* __restrict__ q1h, _Float16* __restrict__ k1h,
    _Float16* __restrict__ q2h, _Float16* __restrict__ k2h,
    _Float16* __restrict__ vTh)
{
  __shared__ _Float16 Xl[128*40];   // [m][k] 128x32, row stride 40 halfs
  __shared__ _Float16 Wl[128*40];   // [n][k] 128x32 (W^T), row stride 40

  const int tid  = threadIdx.x;
  const int lane = tid & 31;
  const int wave = tid >> 5;
  const int n0 = blockIdx.x * 128;
  const int m0 = blockIdx.y * 128;
  const int moff = (wave & 3) * 32;   // wave covers 32 rows x 64 cols
  const int noff = (wave >> 2) * 64;
  const int am = lane & 15, kg = lane >> 4;

  const float* wptr[5] = {wq1, wk1, wq2, wk2, wv};

  v8f acc[2][4];
#pragma unroll
  for (int i = 0; i < 2; ++i)
#pragma unroll
    for (int j = 0; j < 4; ++j) acc[i][j] = vzero8();

  // staging maps
  const int xrow = tid >> 1;            // 0..127
  const int xcol = (tid & 1) * 16;      // 0 / 16
  const int wk   = tid & 31;            // k_local 0..31
  const int wns  = tid >> 5;            // n segment 0..7 (16 cols each)
  const int nbase = n0 + wns * 16;
  const int wsel  = nbase >> 10;
  const int hsel  = (nbase >> 6) & 15;
  const int dbase = nbase & 63;
  const float* wsrc = wptr[wsel] + (size_t)hsel * C_ * D_ + dbase;

  for (int k0 = 0; k0 < C_; k0 += 32) {
    { // stage X tile (fp32 -> f16)
      const float* xp = x + (size_t)(m0 + xrow) * C_ + k0 + xcol;
      if (k0 + 32 < C_) __builtin_prefetch(xp + 32, 0, 3);   // global_prefetch next slice
      _Float16* dst = &Xl[xrow*40 + xcol];
#pragma unroll
      for (int i = 0; i < 4; ++i) {
        float4 a = *(const float4*)(xp + i*4);
        dst[i*4+0] = (_Float16)a.x; dst[i*4+1] = (_Float16)a.y;
        dst[i*4+2] = (_Float16)a.z; dst[i*4+3] = (_Float16)a.w;
      }
    }
    { // stage W^T tile: Wl[n][k] = W[c=k][n]
      const float* wp = wsrc + (size_t)(k0 + wk) * D_;
      float4 f0 = *(const float4*)(wp);
      float4 f1 = *(const float4*)(wp + 4);
      float4 f2 = *(const float4*)(wp + 8);
      float4 f3 = *(const float4*)(wp + 12);
      float f[16] = {f0.x,f0.y,f0.z,f0.w, f1.x,f1.y,f1.z,f1.w,
                     f2.x,f2.y,f2.z,f2.w, f3.x,f3.y,f3.z,f3.w};
#pragma unroll
      for (int i = 0; i < 16; ++i)
        Wl[(wns*16 + i)*40 + wk] = (_Float16)f[i];
    }
    __syncthreads();

    v16h afrag[2], bfrag[4];
#pragma unroll
    for (int mi = 0; mi < 2; ++mi) {
      const _Float16* ap = &Xl[(moff + mi*16 + am)*40];
      afrag[mi] = cat8(*(const v8h*)(ap + kg*8), *(const v8h*)(ap + 16 + kg*8));
    }
#pragma unroll
    for (int ni = 0; ni < 4; ++ni)
        bfrag[ni] = cat8(*(const v8h*)&Wl[(noff + ni*16 + am)*40 + kg*8], *(const v8h*)&Wl[(noff + ni*16 + am)*40 + 16 + kg*8]);
#pragma unroll
    for (int mi = 0; mi < 2; ++mi)
#pragma unroll
      for (int ni = 0; ni < 4; ++ni)
        acc[mi][ni] = WMMA_F16(afrag[mi], bfrag[ni], acc[mi][ni]);
    __syncthreads();
  }

  _Float16* outp[5] = {q1h, k1h, q2h, k2h, vTh};
#pragma unroll
  for (int mi = 0; mi < 2; ++mi) {
#pragma unroll
    for (int ni = 0; ni < 4; ++ni) {
      const int nt = n0 + noff + ni*16;
      const int w  = nt >> 10;
      const int h  = (nt >> 6) & 15;
      const int d  = (nt & 63) + am;
      _Float16* op = outp[w];
#pragma unroll
      for (int r = 0; r < 8; ++r) {
        const int mg = m0 + moff + mi*16 + kg*8 + r;   // C/D layout: row = (lane>>4)*8+r
        const int b  = mg >> 11;
        const int t  = mg & (T_ - 1);
        size_t off;
        if (w < 4) off = (((size_t)(b*H_ + h))*T_ + t)*D_ + d;
        else       off = (((size_t)(b*H_ + h))*D_ + d)*T_ + t;  // v transposed
        op[off] = (_Float16)acc[mi][ni][r];
      }
    }
  }
}

// ---------------------------------------------------------------------------
// Flash-style online-softmax update for one 128x64 score block (per wave:
// 16 rows x 64 cols in 4 acc tiles). Writes P (f16) through a per-wave LDS
// strip to convert acc layout -> A-fragment layout, then O += P @ V via WMMA.
// ---------------------------------------------------------------------------
static __device__ __forceinline__ void softmax_flash_update(
    v8f sacc[4], float* mst, float* lst, v8f oacc[4],
    _Float16* Pl, const _Float16* Vl, int lane, int wave)
{
  const int am = lane & 15, kg = lane >> 4;
  float mnew[8], corr[8], lsum[8];
#pragma unroll
  for (int r = 0; r < 8; ++r) {
    float v = fmaxf(fmaxf(sacc[0][r], sacc[1][r]), fmaxf(sacc[2][r], sacc[3][r]));
#pragma unroll
    for (int off = 1; off < 16; off <<= 1)
      v = fmaxf(v, __shfl_xor(v, off, 32));   // 16-lane row reduce
    mnew[r] = fmaxf(mst[r], v);
    corr[r] = __expf(mst[r] - mnew[r]);
    mst[r]  = mnew[r];
    lsum[r] = 0.f;
  }
#pragma unroll
  for (int dt = 0; dt < 4; ++dt)
#pragma unroll
    for (int r = 0; r < 8; ++r) oacc[dt][r] *= corr[r];

  _Float16* prow = Pl + wave*16*40;   // per-wave strip [16][32+pad]
#pragma unroll
  for (int ks = 0; ks < 2; ++ks) {    // two K=32 chunks of the 64 s-columns
#pragma unroll
    for (int tt = 0; tt < 2; ++tt) {
      const int nt = ks*2 + tt;
#pragma unroll
      for (int r = 0; r < 8; ++r) {
        float p = __expf(sacc[nt][r] - mnew[r]);
        lsum[r] += p;
        prow[(kg*8 + r)*40 + tt*16 + am] = (_Float16)p;
      }
    }
    // gather A-fragment for this K-chunk (same-wave LDS RAW is in-order)
    v8h lo = *(const v8h*)(prow + am*40 + kg*8);
    v8h hi = *(const v8h*)(prow + am*40 + 16 + kg*8);
    v16h af = cat8(lo, hi);
#pragma unroll
    for (int dt = 0; dt < 4; ++dt) {
      v16h bf = cat8(*(const v8h*)(Vl + (dt*16 + am)*72 + ks*32 + kg*8), *(const v8h*)(Vl + (dt*16 + am)*72 + ks*32 + 16 + kg*8));
      oacc[dt] = WMMA_F16(af, bf, oacc[dt]);
    }
  }
#pragma unroll
  for (int r = 0; r < 8; ++r) {
    float v = lsum[r];
#pragma unroll
    for (int off = 1; off < 16; off <<= 1)
      v += __shfl_xor(v, off, 32);
    lst[r] = lst[r]*corr[r] + v;
  }
}

// ---------------------------------------------------------------------------
// Kernel 2: differential attention, flash style. One block = (b, h, 128 t's).
// out_pre_ln[b,t,h*64+d] = softmax(q1 k1^T/8) V - lamb * softmax(q2 k2^T/8) V
// K/V tiles staged via the Tensor Data Mover when available.
// ---------------------------------------------------------------------------
__global__ __launch_bounds__(256) void diff_attn_kernel(
    const _Float16* __restrict__ q1h, const _Float16* __restrict__ k1h,
    const _Float16* __restrict__ q2h, const _Float16* __restrict__ k2h,
    const _Float16* __restrict__ vTh, const float* __restrict__ lambp,
    float* __restrict__ aout)
{
  __shared__ _Float16 Kl[64*72];     // K tile [s][d], reused k1 then k2
  __shared__ _Float16 Vl[64*72];     // V^T tile [d][s]
  __shared__ _Float16 Pl[128*40];    // P transpose strips, per wave

  const int tid = threadIdx.x, lane = tid & 31, wave = tid >> 5;
  const int tb = blockIdx.x & 15;
  const int h  = (blockIdx.x >> 4) & 15;
  const int b  = blockIdx.x >> 8;
  const int t0 = tb * 128;
  const size_t bh = (size_t)(b*H_ + h);
  const int am = lane & 15, kg = lane >> 4;

  // q fragments resident in registers (A layout, straight from global)
  v16h q1f[2], q2f[2];
#pragma unroll
  for (int ks = 0; ks < 2; ++ks) {
    const _Float16* p1 = q1h + (bh*T_ + t0 + wave*16 + am)*D_ + ks*32 + kg*8;
    q1f[ks] = cat8(*(const v8h*)p1, *(const v8h*)(p1 + 16));
    const _Float16* p2 = q2h + (bh*T_ + t0 + wave*16 + am)*D_ + ks*32 + kg*8;
    q2f[ks] = cat8(*(const v8h*)p2, *(const v8h*)(p2 + 16));
  }

  v8f o1[4], o2[4];
  float m1[8], l1[8], m2[8], l2[8];
#pragma unroll
  for (int dt = 0; dt < 4; ++dt) { o1[dt] = vzero8(); o2[dt] = vzero8(); }
#pragma unroll
  for (int r = 0; r < 8; ++r) { m1[r] = -1e30f; m2[r] = -1e30f; l1[r] = 0.f; l2[r] = 0.f; }

#if USE_TDM
  const unsigned klOff = (unsigned)(uintptr_t)&Kl[0];
  const unsigned vlOff = (unsigned)(uintptr_t)&Vl[0];
#else
  const int srow = tid >> 2;          // 0..63
  const int sseg = (tid & 3) * 16;    // 0,16,32,48
#endif

  for (int s0 = 0; s0 < T_; s0 += 64) {
    // ---- stage k1 tile and V^T tile ----
#if USE_TDM
    if (wave == 0) {
      tdm_load_2d_f16(klOff, k1h + (bh*T_ + s0)*D_, D_, 64, D_);
      tdm_load_2d_f16(vlOff, vTh + bh*(size_t)D_*T_ + s0, 64, D_, T_);
      __builtin_amdgcn_s_wait_tensorcnt(0);
    }
#else
    *(v16h*)&Kl[srow*72 + sseg] = *(const v16h*)(k1h + (bh*T_ + s0 + srow)*D_ + sseg);
    *(v16h*)&Vl[srow*72 + sseg] = *(const v16h*)(vTh + (bh*D_ + srow)*T_ + s0 + sseg);
#endif
    __syncthreads();

    v8f s1[4];
#pragma unroll
    for (int nt = 0; nt < 4; ++nt) {
      s1[nt] = vzero8();
#pragma unroll
      for (int ks = 0; ks < 2; ++ks) {
        v16h bf = cat8(*(const v8h*)&Kl[(nt*16 + am)*72 + ks*32 + kg*8], *(const v8h*)&Kl[(nt*16 + am)*72 + ks*32 + 16 + kg*8]);
        s1[nt] = WMMA_F16(q1f[ks], bf, s1[nt]);
      }
    }
    __syncthreads();
    // ---- stage k2 into Kl (S1 done); overlap with attn1 softmax/O-update ----
#if USE_TDM
    if (wave == 0)
      tdm_load_2d_f16(klOff, k2h + (bh*T_ + s0)*D_, D_, 64, D_);
#else
    *(v16h*)&Kl[srow*72 + sseg] = *(const v16h*)(k2h + (bh*T_ + s0 + srow)*D_ + sseg);
#endif

#pragma unroll
    for (int nt = 0; nt < 4; ++nt)
#pragma unroll
      for (int r = 0; r < 8; ++r) s1[nt][r] *= 0.125f;   // 1/sqrt(64)
    softmax_flash_update(s1, m1, l1, o1, Pl, Vl, lane, wave);
#if USE_TDM
    if (wave == 0) __builtin_amdgcn_s_wait_tensorcnt(0);
#endif
    __syncthreads();

    v8f s2[4];
#pragma unroll
    for (int nt = 0; nt < 4; ++nt) {
      s2[nt] = vzero8();
#pragma unroll
      for (int ks = 0; ks < 2; ++ks) {
        v16h bf = cat8(*(const v8h*)&Kl[(nt*16 + am)*72 + ks*32 + kg*8], *(const v8h*)&Kl[(nt*16 + am)*72 + ks*32 + 16 + kg*8]);
        s2[nt] = WMMA_F16(q2f[ks], bf, s2[nt]);
      }
    }
#pragma unroll
    for (int nt = 0; nt < 4; ++nt)
#pragma unroll
      for (int r = 0; r < 8; ++r) s2[nt][r] *= 0.125f;
    softmax_flash_update(s2, m2, l2, o2, Pl, Vl, lane, wave);
    __syncthreads();   // protect Kl/Vl before next iteration's staging
  }

  const float lamb = *lambp;
#pragma unroll
  for (int dt = 0; dt < 4; ++dt) {
#pragma unroll
    for (int r = 0; r < 8; ++r) {
      const int t = t0 + wave*16 + kg*8 + r;
      const int d = dt*16 + am;
      float v = o1[dt][r]/l1[r] - lamb*(o2[dt][r]/l2[r]);
      aout[((size_t)b*T_ + t)*C_ + h*D_ + d] = v;
    }
  }
}

// ---------------------------------------------------------------------------
// Kernel 3: LayerNorm over C=1024 per (b,t) row, then scale by (1-lamb).
// ---------------------------------------------------------------------------
__global__ __launch_bounds__(256) void layernorm_kernel(
    const float* __restrict__ ain, const float* __restrict__ gamma,
    const float* __restrict__ beta, const float* __restrict__ lambp,
    float* __restrict__ out)
{
  __shared__ float reds[8], reds2[8], stats[2];
  const int row = blockIdx.x, tid = threadIdx.x, lane = tid & 31, wave = tid >> 5;
  const float4 v = *(const float4*)(ain + (size_t)row*C_ + tid*4);
  float s  = v.x + v.y + v.z + v.w;
  float s2 = v.x*v.x + v.y*v.y + v.z*v.z + v.w*v.w;
#pragma unroll
  for (int off = 1; off < 32; off <<= 1) {
    s  += __shfl_xor(s,  off, 32);
    s2 += __shfl_xor(s2, off, 32);
  }
  if (lane == 0) { reds[wave] = s; reds2[wave] = s2; }
  __syncthreads();
  if (tid == 0) {
    float a = 0.f, c = 0.f;
#pragma unroll
    for (int i = 0; i < 8; ++i) { a += reds[i]; c += reds2[i]; }
    stats[0] = a; stats[1] = c;
  }
  __syncthreads();
  const float mu  = stats[0] * (1.0f/C_);
  const float var = stats[1] * (1.0f/C_) - mu*mu;
  const float inv = rsqrtf(var + 1e-5f);
  const float oml = 1.0f - *lambp;
  const float4 g  = *(const float4*)(gamma + tid*4);
  const float4 bb = *(const float4*)(beta  + tid*4);
  float4 o;
  o.x = ((v.x - mu)*inv*g.x + bb.x)*oml;
  o.y = ((v.y - mu)*inv*g.y + bb.y)*oml;
  o.z = ((v.z - mu)*inv*g.z + bb.z)*oml;
  o.w = ((v.w - mu)*inv*g.w + bb.w)*oml;
  *(float4*)(out + (size_t)row*C_ + tid*4) = o;
}

// ---------------------------------------------------------------------------
extern "C" void kernel_launch(void* const* d_in, const int* in_sizes, int n_in,
                              void* d_out, int out_size, void* d_ws, size_t ws_size,
                              hipStream_t stream)
{
  const float* x    = (const float*)d_in[0];
  const float* wq1  = (const float*)d_in[1];
  const float* wk1  = (const float*)d_in[2];
  const float* wq2  = (const float*)d_in[3];
  const float* wk2  = (const float*)d_in[4];
  const float* wv   = (const float*)d_in[5];
  const float* g    = (const float*)d_in[6];
  const float* bta  = (const float*)d_in[7];
  const float* lamb = (const float*)d_in[8];

  const size_t nElem = (size_t)B_ * H_ * T_ * D_;   // 4,194,304 per tensor
  char* ws = (char*)d_ws;
  _Float16* q1h = (_Float16*)ws;
  _Float16* k1h = q1h + nElem;
  _Float16* q2h = k1h + nElem;
  _Float16* k2h = q2h + nElem;
  _Float16* vTh = k2h + nElem;
  float* aout = (float*)(ws + 5 * nElem * sizeof(_Float16));

  dim3 gproj(NCOL_/128, (B_*T_)/128);   // 40 x 32
  proj_gemm_kernel<<<gproj, 256, 0, stream>>>(x, wq1, wk1, wq2, wk2, wv,
                                              q1h, k1h, q2h, k2h, vTh);
  diff_attn_kernel<<<B_*H_*(T_/128), 256, 0, stream>>>(q1h, k1h, q2h, k2h, vTh,
                                                       lamb, aout);
  layernorm_kernel<<<B_*T_, 256, 0, stream>>>(aout, g, bta, lamb, (float*)d_out);
}
